// ClassKNN_74749610820316
// MI455X (gfx1250) — matrix operand order repaired and checked by reading, not compiled, not run on hardware
//
#include <hip/hip_runtime.h>
#include <math.h>

typedef __attribute__((ext_vector_type(16))) _Float16 v16h;
typedef __attribute__((ext_vector_type(8)))  _Float16 v8h;
typedef __attribute__((ext_vector_type(8)))  float    v8f;

#define BTOT 16384
// d_out layout (floats): u_hat[0,32768) u_col[32768,65536) A2[65536,131072) b2[131072,163840)
#define OFF_UCOL 32768
#define OFF_A2   65536
#define OFF_B2   131072
// packed weights per MLP (halves): L0 (128x32)=4096, L1/L2 (128x128)=16384, L3 (16x128)=2048
#define MLP_W_HALVES 38912
#define W_L0 0
#define W_L1 4096
#define W_L2 20480
#define W_L3 36864

__device__ __forceinline__ v16h cat8(v8h lo, v8h hi) {
  return __builtin_shufflevector(lo, hi, 0,1,2,3,4,5,6,7,8,9,10,11,12,13,14,15);
}

// One MLP layer for one wave: 16 rows, dout = NT*16 cols, K padded to KP.
// aIn: LDS, 16 rows x asz halves (row-major). aOut: LDS, 16 rows x 128 halves.
// wT: global f16, (NT*16) rows x KP halves (transposed weights, zero-padded).
// ACT: 0 none, 1 tanh, 2 elu. In-place (aIn==aOut) is safe: all loads precede stores.
template<int KP, int NT, int ACT>
__device__ __forceinline__ void mlp_layer(const _Float16* aIn, int asz, _Float16* aOut,
                                          const _Float16* __restrict__ wT,
                                          const float* __restrict__ bias)
{
  const int lane  = threadIdx.x & 31;
  const int half_ = lane >> 4;
  const int l16   = lane & 15;
  v8f accs[NT];
  #pragma unroll
  for (int nt = 0; nt < NT; ++nt) {
    v8f acc = {};
    #pragma unroll
    for (int kc = 0; kc < KP / 32; ++kc) {
      // A fragment (16x32 f16): lanes<16: K {0..7,16..23}; lanes>=16: K {8..15,24..31}
      const _Float16* ap = aIn + l16 * asz + kc * 32 + half_ * 8;
      v16h a = cat8(*(const v8h*)ap, *(const v8h*)(ap + 16));
      // B fragment (32x16 f16): N=l16; lanes<16: K {0..7,16..23}; lanes>=16: K {8..15,24..31}
      const _Float16* bpp = wT + (nt * 16 + l16) * KP + kc * 32 + half_ * 8;
      v16h b = cat8(*(const v8h*)bpp, *(const v8h*)(bpp + 16));
      acc = __builtin_amdgcn_wmma_f32_16x16x32_f16(false, a, false, b, (short)0, acc,
                                                   false, false);
    }
    #pragma unroll
    for (int r = 0; r < 8; ++r) {
      float v = acc[r] + bias[nt * 16 + l16];
      if (ACT == 1)      v = tanhf(v);
      else if (ACT == 2) v = (v > 0.f) ? v : (__expf(v) - 1.f);
      acc[r] = v;
    }
    accs[nt] = acc;
  }
  // C layout: element r of lane -> row M = r + 8*half_, col N = nt*16 + l16
  #pragma unroll
  for (int nt = 0; nt < NT; ++nt)
    #pragma unroll
    for (int r = 0; r < 8; ++r)
      aOut[(r + half_ * 8) * 128 + nt * 16 + l16] = (_Float16)accs[nt][r];
}

// Final layer: single 16x16 tile, bias added, no activation, returned in registers.
template<int KP>
__device__ __forceinline__ v8f mlp_layer_final(const _Float16* aIn,
                                               const _Float16* __restrict__ wT,
                                               const float* __restrict__ bias)
{
  const int lane  = threadIdx.x & 31;
  const int half_ = lane >> 4;
  const int l16   = lane & 15;
  v8f acc = {};
  #pragma unroll
  for (int kc = 0; kc < KP / 32; ++kc) {
    const _Float16* ap = aIn + l16 * 128 + kc * 32 + half_ * 8;
    v16h a = cat8(*(const v8h*)ap, *(const v8h*)(ap + 16));
    const _Float16* bpp = wT + l16 * KP + kc * 32 + half_ * 8;
    v16h b = cat8(*(const v8h*)bpp, *(const v8h*)(bpp + 16));
    acc = __builtin_amdgcn_wmma_f32_16x16x32_f16(false, a, false, b, (short)0, acc,
                                                 false, false);
  }
  #pragma unroll
  for (int r = 0; r < 8; ++r) acc[r] += bias[l16];
  return acc;
}

// ---- weight packing: W (din,dout) f32 row-major -> wT (NP,KP) f16 transposed, zero padded
__global__ void pack_layer(const float* __restrict__ W, const float* __restrict__ bias,
                           _Float16* __restrict__ wT, float* __restrict__ bOut,
                           int din, int dout, int KP, int NP)
{
  int i = blockIdx.x * blockDim.x + threadIdx.x;
  if (i < KP * NP) {
    int n = i / KP, k = i % KP;
    wT[i] = (_Float16)((k < din && n < dout) ? W[k * dout + n] : 0.f);
  }
  if (i < NP) bOut[i] = (i < dout) ? bias[i] : 0.f;
}

// ---- elementwise prep: psi0, Lfb0; also writes u_col, constant A2, b2[:,1]
__global__ void prep_kernel(const float* __restrict__ x, const float* __restrict__ u,
                            const float* __restrict__ xo, const float* __restrict__ ro,
                            float* __restrict__ psi0, float* __restrict__ lfb0,
                            float* __restrict__ out, int B)
{
  int b = blockIdx.x * blockDim.x + threadIdx.x;
  if (b >= B) return;
  float x0 = x[b*4+0], x1 = x[b*4+1], x2 = x[b*4+2], x3 = x[b*4+3];
  float dx = x0 - xo[0], dy = x1 - xo[1], r = ro[0];
  psi0[b] = dx*dx + dy*dy - r*r;
  lfb0[b] = 2.f*dx*x2*cosf(x3) + 2.f*dy*x2*sinf(x3);
  float* ucol = out + OFF_UCOL;
  float* A2   = out + OFF_A2;
  float* b2   = out + OFF_B2;
  ucol[2*b]   = u[2*b];
  ucol[2*b+1] = u[2*b+1];
  A2[4*b+0] = 0.f; A2[4*b+1] = 0.f; A2[4*b+2] = 1.f; A2[4*b+3] = 0.f; // [[-Lgb0],[-Lgb1]]
  b2[2*b+1] = 10.f - x2;                                              // b1 = psi01
}

// ---- policy (MODE 0, tanh) / scale (MODE 1, elu) MLP forward; 4 waves x 16-row tiles
template<int MODE>
__global__ __launch_bounds__(128) void mlp_fwd_kernel(const float* __restrict__ x,
    const float* __restrict__ xo, const _Float16* __restrict__ wp,
    const float* __restrict__ bp, float* __restrict__ out0, float* __restrict__ out1)
{
  __shared__ __align__(16) _Float16 sIn[4][16 * 32];
  __shared__ __align__(16) _Float16 sAct[4][16 * 128];
  const int w = threadIdx.x >> 5, lane = threadIdx.x & 31;
  const int half_ = lane >> 4, l16 = lane & 15;
  const int rb = blockIdx.x * 64 + w * 16;
  for (int i = lane; i < 512; i += 32) {          // stage 16x32 padded inputs
    int k = i & 31, m = i >> 5;
    float v = 0.f;
    if (k < 4)                    v = x[(rb + m) * 4 + k];
    else if (MODE == 1 && k == 4) v = xo[0];
    else if (MODE == 1 && k == 5) v = xo[1];
    sIn[w][i] = (_Float16)v;
  }
  constexpr int ACT = (MODE == 0) ? 1 : 2;
  mlp_layer<32, 8, ACT>(sIn[w], 32, sAct[w], wp + W_L0, bp + 0);
  mlp_layer<128, 8, ACT>(sAct[w], 128, sAct[w], wp + W_L1, bp + 128);
  mlp_layer<128, 8, ACT>(sAct[w], 128, sAct[w], wp + W_L2, bp + 256);
  v8f o = mlp_layer_final<128>(sAct[w], wp + W_L3, bp + 384);
  if (MODE == 0) {                                 // u_nn: cols 0,1
    if (l16 < 2) {
      #pragma unroll
      for (int r = 0; r < 8; ++r) out0[(rb + half_ * 8 + r) * 2 + l16] = o[r];
    }
  } else {                                         // offset = v0, scaling = exp(v1)
    if (l16 == 0) {
      #pragma unroll
      for (int r = 0; r < 8; ++r) out0[rb + half_ * 8 + r] = o[r];
    } else if (l16 == 1) {
      #pragma unroll
      for (int r = 0; r < 8; ++r) out1[rb + half_ * 8 + r] = expf(o[r]);
    }
  }
}

// ---- int MLP over 51 steps, trapezoid-integrated. One block per 16 batch rows.
__global__ __launch_bounds__(128) void int_kernel(const float* __restrict__ x,
    const float* __restrict__ xo, const _Float16* __restrict__ wp,
    const float* __restrict__ bp, const float* __restrict__ psi0,
    const float* __restrict__ lfb0, const float* __restrict__ offs,
    const float* __restrict__ scal, float* __restrict__ b0out, float* __restrict__ b2out)
{
  __shared__ __align__(16) _Float16 sIn[4][16 * 32];
  __shared__ __align__(16) _Float16 sAct[4][16 * 128];
  __shared__ float sPart[4][16];
  const int w = threadIdx.x >> 5, lane = threadIdx.x & 31;
  const int half_ = lane >> 4, l16 = lane & 15;
  const int rb = blockIdx.x * 16;
  for (int i = lane; i < 512; i += 32) {          // static features: [t, x0..x3, xo0, xo1]
    int k = i & 31, m = i >> 5;
    float v = 0.f;
    if (k >= 1 && k < 5) v = x[(rb + m) * 4 + (k - 1)];
    else if (k == 5)     v = xo[0];
    else if (k == 6)     v = xo[1];
    sIn[w][i] = (_Float16)v;
  }
  const float psi_m = psi0[rb + l16];
  v8f accv = {};
  #pragma unroll 1
  for (int j = 0; j < 13; ++j) {                  // wave w handles s = w + 4j
    int s = w + 4 * j;
    float wgt = (s == 0 || s == 50) ? 0.01f : ((s < 51) ? 0.02f : 0.f);
    int sc = (s < 51) ? s : 50;
    if (lane < 16) sIn[w][l16 * 32] = (_Float16)(psi_m * (float)sc * 0.02f); // t column
    mlp_layer<32, 8, 2>(sIn[w], 32, sAct[w], wp + W_L0, bp + 0);
    mlp_layer<128, 8, 2>(sAct[w], 128, sAct[w], wp + W_L1, bp + 128);
    mlp_layer<128, 8, 2>(sAct[w], 128, sAct[w], wp + W_L2, bp + 256);
    v8f o = mlp_layer_final<128>(sAct[w], wp + W_L3, bp + 384);
    #pragma unroll
    for (int r = 0; r < 8; ++r) {                 // f = elu(z)+1, column N=0 (lanes 0/16)
      float z = o[r];
      float f = ((z > 0.f) ? z : (__expf(z) - 1.f)) + 1.f;
      accv[r] += wgt * f;
    }
  }
  if (l16 == 0) {                                 // lanes 0 (M0..7) and 16 (M8..15)
    #pragma unroll
    for (int r = 0; r < 8; ++r) sPart[w][half_ * 8 + r] = accv[r];
  }
  __syncthreads();
  if (threadIdx.x < 16) {
    int m = threadIdx.x, b = rb + m;
    float integral = sPart[0][m] + sPart[1][m] + sPart[2][m] + sPart[3][m];
    float mono = scal[b] * (psi0[b] * integral) + offs[b];
    float bb0 = lfb0[b] + mono;
    b0out[b] = bb0;
    b2out[2 * b] = bb0;
  }
}

// ---- QP dual ascent: G rows = [0,0],[1,0],[1,0],[0,1],[-1,0],[0,-1]; h=[b0,b1,5,2,5,2]
__global__ void qp_kernel(const float* __restrict__ unn, const float* __restrict__ b0v,
                          const float* __restrict__ x, float* __restrict__ uhat, int B)
{
  int b = blockIdx.x * blockDim.x + threadIdx.x;
  if (b >= B) return;
  float u0 = unn[2*b], u1 = unn[2*b+1];
  float h0 = b0v[b], h1 = 10.f - x[b*4+2];
  float l0 = 0.f, l1 = 0.f, l2 = 0.f, l3 = 0.f, l4 = 0.f, l5 = 0.f;
  #pragma unroll 1
  for (int it = 0; it < 100; ++it) {
    float uu0 = u0 - (l1 + l2 - l4);
    float uu1 = u1 - (l3 - l5);
    l0 = fmaxf(0.f, l0 + 0.05f * (0.f  - h0));
    l1 = fmaxf(0.f, l1 + 0.05f * (uu0  - h1));
    l2 = fmaxf(0.f, l2 + 0.05f * (uu0  - 5.f));
    l3 = fmaxf(0.f, l3 + 0.05f * (uu1  - 2.f));
    l4 = fmaxf(0.f, l4 + 0.05f * (-uu0 - 5.f));
    l5 = fmaxf(0.f, l5 + 0.05f * (-uu1 - 2.f));
  }
  uhat[2*b]   = u0 - (l1 + l2 - l4);
  uhat[2*b+1] = u1 - (l3 - l5);
}

extern "C" void kernel_launch(void* const* d_in, const int* in_sizes, int n_in,
                              void* d_out, int out_size, void* d_ws, size_t ws_size,
                              hipStream_t stream)
{
  (void)in_sizes; (void)n_in; (void)out_size; (void)ws_size;
  const int B = BTOT;
  const float* x  = (const float*)d_in[0];
  const float* u  = (const float*)d_in[1];
  const float* xo = (const float*)d_in[26];
  const float* ro = (const float*)d_in[27];

  // workspace carve-up
  _Float16* wPack = (_Float16*)d_ws;                                  // 3*38912 halves
  float* bPack = (float*)((char*)d_ws + 3 * MLP_W_HALVES * 2);        // 3*512 floats
  float* psi0 = bPack + 3 * 512;
  float* lfb0 = psi0 + B;
  float* unn  = lfb0 + B;       // B*2
  float* offs = unn + 2 * B;
  float* scal = offs + B;
  float* b0v  = scal + B;

  float* out = (float*)d_out;

  // pack weights (f32 -> f16, transposed, padded): mlp m in {0:policy,1:int,2:scale}
  const int woff[4] = {W_L0, W_L1, W_L2, W_L3};
  for (int m = 0; m < 3; ++m) {
    for (int l = 0; l < 4; ++l) {
      const float* W  = (const float*)d_in[2 + 8 * m + 2 * l];
      const float* bb = (const float*)d_in[2 + 8 * m + 2 * l + 1];
      int din  = (l == 0) ? ((m == 0) ? 4 : (m == 1) ? 7 : 6) : 128;
      int dout = (l == 3) ? ((m == 1) ? 1 : 2) : 128;
      int KP = (l == 0) ? 32 : 128;
      int NP = (l == 3) ? 16 : 128;
      int tot = KP * NP;
      pack_layer<<<(tot + 255) / 256, 256, 0, stream>>>(
          W, bb, wPack + m * MLP_W_HALVES + woff[l], bPack + m * 512 + l * 128,
          din, dout, KP, NP);
    }
  }

  prep_kernel<<<B / 256, 256, 0, stream>>>(x, u, xo, ro, psi0, lfb0, out, B);
  mlp_fwd_kernel<0><<<B / 64, 128, 0, stream>>>(x, xo, wPack + 0 * MLP_W_HALVES,
                                                bPack + 0 * 512, unn, nullptr);
  mlp_fwd_kernel<1><<<B / 64, 128, 0, stream>>>(x, xo, wPack + 2 * MLP_W_HALVES,
                                                bPack + 2 * 512, offs, scal);
  int_kernel<<<B / 16, 128, 0, stream>>>(x, xo, wPack + 1 * MLP_W_HALVES,
                                         bPack + 1 * 512, psi0, lfb0, offs, scal,
                                         b0v, out + OFF_B2);
  qp_kernel<<<B / 256, 256, 0, stream>>>(unn, b0v, x, out, B);
}
